// Attention_12773232739032
// MI455X (gfx1250) — matrix operand order repaired and checked by reading, not compiled, not run on hardware
//
#include <hip/hip_runtime.h>

#define NUM_HEADS    32
#define NUM_KV_HEADS 8
#define HEAD_DIM     128
#define GQA          4
#define ATTN_SCALE   0.08838834764831845f  // 1/sqrt(128)

typedef __attribute__((ext_vector_type(16))) _Float16 v16h;
typedef __attribute__((ext_vector_type(8)))  float    v8f;

// ---------------------------------------------------------------------------
// CDNA5 async global->LDS DMA (ASYNCcnt-tracked), GV addressing mode.
// ---------------------------------------------------------------------------
__device__ __forceinline__ void async_load_b128_to_lds(unsigned lds_byte_addr,
                                                       const float* gaddr) {
  asm volatile("global_load_async_to_lds_b128 %0, %1, off"
               :: "v"(lds_byte_addr), "v"(gaddr) : "memory");
}
__device__ __forceinline__ void wait_async0() {
  asm volatile("s_wait_asynccnt 0x0" ::: "memory");
}
__device__ __forceinline__ void wait_ds0() {
  asm volatile("s_wait_dscnt 0x0" ::: "memory");
}

// ---------------------------------------------------------------------------
// KV-cache scatter: k_cache_new[slot] = k[t], v_cache_new[slot] = v[t]
// (base cache contents are copied in via hipMemcpyAsync before this runs)
// ---------------------------------------------------------------------------
__global__ void kv_scatter_kernel(const float* __restrict__ k,
                                  const float* __restrict__ v,
                                  const int* __restrict__ slot_mapping,
                                  float* __restrict__ kc,
                                  float* __restrict__ vc,
                                  int total) {
  const int row_elems = NUM_KV_HEADS * HEAD_DIM;
  int tid = blockIdx.x * blockDim.x + threadIdx.x;
  int n = total * row_elems;
  if (tid >= n) return;
  int t   = tid / row_elems;
  int rem = tid - t * row_elems;
  int slot = slot_mapping[t];
  if (slot < 0) return;
  size_t dst = (size_t)slot * row_elems + rem;
  kc[dst] = k[tid];
  vc[dst] = v[tid];
}

// ---------------------------------------------------------------------------
// Varlen causal GQA flash attention.
// One wave32 per (16-row query tile, head). f16 WMMA, f32 accumulation.
// K/V staged by double-buffered async DMA; next block's DMA overlaps compute.
// Dynamic LDS layout (65 KB):
//   [0,16K)  kbuf0   [16K,32K) kbuf1   [32K,48K) vbuf0   [48K,64K) vbuf1
//   [64K,65K) lds_p (f16 16x32)
// ---------------------------------------------------------------------------
__global__ __launch_bounds__(32) void attn_prefill_kernel(
    const float* __restrict__ q, const float* __restrict__ k,
    const float* __restrict__ v, const int* __restrict__ cu_q,
    const int* __restrict__ cu_k, float* __restrict__ out) {
  extern __shared__ __align__(16) char smem_raw[];
  // NOTE: compute buffer pointers arithmetically at runtime; a const array of
  // pointers into LDS becomes a static addrspacecast initializer ld.lld rejects.
  char* sm = smem_raw;
  _Float16* lds_p = (_Float16*)(sm + 65536);

  const int b    = blockIdx.z;
  const int h    = blockIdx.y;
  const int qt   = blockIdx.x;
  const int lane = threadIdx.x;
  const int half = lane >> 4;   // 0: lanes 0-15, 1: lanes 16-31
  const int lm   = lane & 15;

  const int q_start = cu_q[b];
  const int len     = cu_q[b + 1] - q_start;
  const int k_start = cu_k[b];
  const int len_k   = cu_k[b + 1] - k_start;
  const int q0      = qt * 16;
  if (q0 >= len) return;
  const int kvh = h / GQA;

  // ---- Load Q tile into WMMA A-layout (M = lane%16; K split by half) ----
  v16h qa[4];
  {
    int row  = q0 + lm;
    int rowc = (row < len) ? row : (len - 1);
    const float* qrow = q + ((size_t)(q_start + rowc) * NUM_HEADS + h) * HEAD_DIM;
    float s = (row < len) ? ATTN_SCALE : 0.0f;  // zero invalid rows
#pragma unroll
    for (int c = 0; c < 4; ++c) {
#pragma unroll
      for (int vv = 0; vv < 8; ++vv) {
        int kk = c * 32 + (vv & 3) * 2 + (vv >> 2) * 16 + (half ? 8 : 0);
        float2 f = *(const float2*)(qrow + kk);
        qa[c][2 * vv]     = (_Float16)(f.x * s);
        qa[c][2 * vv + 1] = (_Float16)(f.y * s);
      }
    }
  }

  v8f o[8];
#pragma unroll
  for (int t = 0; t < 8; ++t) o[t] = {};
  float mrow[8], lrow[8];
#pragma unroll
  for (int e = 0; e < 8; ++e) { mrow[e] = -1e30f; lrow[e] = 0.0f; }

  int kmax = q0 + 15;
  if (kmax > len_k - 1) kmax = len_k - 1;
  const int nblk = kmax / 32 + 1;

  // Async-DMA one 32-key block (K + V, f32 raw) into buffer `sel`.
  // One b128 async op per 128-float row: 32 lanes x 16B.
  auto issue_block = [&](int j, int sel) {
    float* kb = (float*)(sm + sel * 16384);
    float* vb = (float*)(sm + 32768 + sel * 16384);
    const int lo = lane * 4;
    for (int r = 0; r < 32; ++r) {
      int key  = j + r;
      int keyc = (key < len_k) ? key : (len_k - 1);
      size_t base = ((size_t)(k_start + keyc) * NUM_KV_HEADS + kvh) * HEAD_DIM + lo;
      async_load_b128_to_lds((unsigned)(size_t)(kb + r * HEAD_DIM + lo), k + base);
      async_load_b128_to_lds((unsigned)(size_t)(vb + r * HEAD_DIM + lo), v + base);
    }
  };

  issue_block(0, 0);  // preload first key block

  for (int jb = 0; jb < nblk; ++jb) {
    const int j   = jb * 32;
    const int sel = jb & 1;

    wait_async0();                    // block jb resident in LDS
    if (jb + 1 < nblk) {
      wait_ds0();                     // old reads of the other buffer done
      issue_block(j + 32, sel ^ 1);   // DMA next block; overlaps compute below
    }
    const float* kb = (const float*)(sm + sel * 16384);
    const float* vb = (const float*)(sm + 32768 + sel * 16384);

    // ---- S = Q * K^T : two 16x16 tiles (keys j..j+15, j+16..j+31) ----
    v8f s0 = {}, s1 = {};
#pragma unroll
    for (int c = 0; c < 4; ++c) {
      v16h kb0, kb1;
#pragma unroll
      for (int vv = 0; vv < 8; ++vv) {
        // B layout (16-bit 32x16): N = lane%16; K = 8*half + s (s<8),
        // 16 + 8*half + (s-8) (s>=8).  Pair over d is contiguous f32.
        int dd = c * 32 + (vv & 3) * 2 + (vv >> 2) * 16 + (half ? 8 : 0);
        float2 p0 = *(const float2*)(kb + lm * HEAD_DIM + dd);
        float2 p1 = *(const float2*)(kb + (16 + lm) * HEAD_DIM + dd);
        kb0[2 * vv] = (_Float16)p0.x; kb0[2 * vv + 1] = (_Float16)p0.y;
        kb1[2 * vv] = (_Float16)p1.x; kb1[2 * vv + 1] = (_Float16)p1.y;
      }
      s0 = __builtin_amdgcn_wmma_f32_16x16x32_f16(false, qa[c], false, kb0,
                                                  (short)0, s0, false, false);
      s1 = __builtin_amdgcn_wmma_f32_16x16x32_f16(false, qa[c], false, kb1,
                                                  (short)0, s1, false, false);
    }

    // ---- Mask + online softmax (C layout: row = e + 8*half, col = lm) ----
#pragma unroll
    for (int e = 0; e < 8; ++e) {
      int qp  = q0 + e + half * 8;
      int kp0 = j + lm;
      int kp1 = j + 16 + lm;
      if (kp0 > qp || kp0 >= len_k) s0[e] = -1e30f;
      if (kp1 > qp || kp1 >= len_k) s1[e] = -1e30f;

      float x = fmaxf(s0[e], s1[e]);
      x = fmaxf(x, __shfl_xor(x, 1, 16));
      x = fmaxf(x, __shfl_xor(x, 2, 16));
      x = fmaxf(x, __shfl_xor(x, 4, 16));
      x = fmaxf(x, __shfl_xor(x, 8, 16));
      float mn = fmaxf(mrow[e], x);
      float sc = __expf(mrow[e] - mn);
      mrow[e] = mn;
      s0[e] = __expf(s0[e] - mn);
      s1[e] = __expf(s1[e] - mn);
      float rs = s0[e] + s1[e];
      rs += __shfl_xor(rs, 1, 16);
      rs += __shfl_xor(rs, 2, 16);
      rs += __shfl_xor(rs, 4, 16);
      rs += __shfl_xor(rs, 8, 16);
      lrow[e] = lrow[e] * sc + rs;
#pragma unroll
      for (int t = 0; t < 8; ++t) o[t][e] *= sc;
    }

    // ---- P (C layout) -> LDS -> A layout ----
#pragma unroll
    for (int e = 0; e < 8; ++e) {
      int row = e + half * 8;
      lds_p[row * 32 + lm]      = (_Float16)s0[e];
      lds_p[row * 32 + 16 + lm] = (_Float16)s1[e];
    }
    __syncthreads();
    v16h pa;
#pragma unroll
    for (int vv = 0; vv < 8; ++vv) {
      int kk = (vv & 3) * 2 + (vv >> 2) * 16 + (half ? 8 : 0);
      pa[2 * vv]     = lds_p[lm * 32 + kk];
      pa[2 * vv + 1] = lds_p[lm * 32 + kk + 1];
    }

    // ---- O += P * V : 8 output d-tiles, K=32 over the key block ----
#pragma unroll
    for (int t = 0; t < 8; ++t) {
      v16h vbm;
#pragma unroll
      for (int vv = 0; vv < 8; ++vv) {
        // B layout: N = lane%16 (d column), K pair over keys (row stride).
        int kk = (vv & 3) * 2 + (vv >> 2) * 16 + (half ? 8 : 0);
        int dd = t * 16 + lm;
        vbm[2 * vv]     = (_Float16)vb[kk * HEAD_DIM + dd];
        vbm[2 * vv + 1] = (_Float16)vb[(kk + 1) * HEAD_DIM + dd];
      }
      o[t] = __builtin_amdgcn_wmma_f32_16x16x32_f16(false, pa, false, vbm,
                                                    (short)0, o[t], false, false);
    }
    __syncthreads();
  }

  // ---- Normalize and store valid rows ----
#pragma unroll
  for (int e = 0; e < 8; ++e) {
    int row = q0 + e + half * 8;
    if (row >= len) continue;
    float inv = (lrow[e] > 0.0f) ? (1.0f / lrow[e]) : 0.0f;
    float* orow = out + ((size_t)(q_start + row) * NUM_HEADS + h) * HEAD_DIM;
#pragma unroll
    for (int t = 0; t < 8; ++t) {
      orow[t * 16 + lm] = o[t][e] * inv;
    }
  }
}

// ---------------------------------------------------------------------------
extern "C" void kernel_launch(void* const* d_in, const int* in_sizes, int n_in,
                              void* d_out, int out_size, void* d_ws, size_t ws_size,
                              hipStream_t stream) {
  const float* q            = (const float*)d_in[0];
  const float* k            = (const float*)d_in[1];
  const float* v            = (const float*)d_in[2];
  const float* k_cache      = (const float*)d_in[3];
  const float* v_cache      = (const float*)d_in[4];
  const int*   slot_mapping = (const int*)d_in[5];
  const int*   cu_q         = (const int*)d_in[6];
  const int*   cu_k         = (const int*)d_in[7];
  // d_in[8] = max_seqlen (unused; tile loop is bounded by cu_seqlens on device)

  const int total     = in_sizes[0] / (NUM_HEADS * HEAD_DIM);
  const int num_slots = in_sizes[3] / (NUM_KV_HEADS * HEAD_DIM);
  const int B         = in_sizes[6] - 1;

  float* out_attn = (float*)d_out;
  float* kc_out   = out_attn + (size_t)total * NUM_HEADS * HEAD_DIM;
  float* vc_out   = kc_out + (size_t)num_slots * NUM_KV_HEADS * HEAD_DIM;

  // Base cache copy (capture-safe async D2D), then scatter new KV rows.
  size_t cache_bytes = (size_t)in_sizes[3] * sizeof(float);
  (void)hipMemcpyAsync(kc_out, k_cache, cache_bytes, hipMemcpyDeviceToDevice, stream);
  (void)hipMemcpyAsync(vc_out, v_cache, cache_bytes, hipMemcpyDeviceToDevice, stream);

  {
    int n = total * NUM_KV_HEADS * HEAD_DIM;
    int blocks = (n + 255) / 256;
    kv_scatter_kernel<<<blocks, 256, 0, stream>>>(k, v, slot_mapping,
                                                  kc_out, vc_out, total);
  }

  {
    dim3 grid((total + 15) / 16, NUM_HEADS, B);
    size_t smem = 65536 + 1024;  // 2x(K,V) f32 double buffers + P tile (f16)
    attn_prefill_kernel<<<grid, 32, smem, stream>>>(q, k, v, cu_q, cu_k, out_attn);
  }
}
